// TriangleAttention_49727131353908
// MI455X (gfx1250) — matrix operand order repaired and checked by reading, not compiled, not run on hardware
//
#include <hip/hip_runtime.h>
#include <hip/hip_bf16.h>

typedef __attribute__((ext_vector_type(16))) __bf16 v16bf;
typedef __attribute__((ext_vector_type(8)))  float  v8f;
typedef __attribute__((ext_vector_type(4)))  unsigned int u32x4;
typedef __attribute__((ext_vector_type(8)))  int  i32x8;
typedef __attribute__((ext_vector_type(4)))  int  i32x4;

#define L 320
#define C 128
#define H 4
#define D 32
#define NQKV 384
#define NCAT 512   // 384 qkv + 128 gate columns fused

__device__ __forceinline__ v8f wmma_bf16(v16bf a, v16bf b, v8f c) {
  // (neg_a, A, neg_b, B, c_mod, C, reuse_a, reuse_b)
  return __builtin_amdgcn_wmma_f32_16x16x32_bf16(false, a, false, b, (short)0, c,
                                                 false, false);
}

__device__ __forceinline__ float rmax16(float x) {
  x = fmaxf(x, __shfl_xor(x, 1));
  x = fmaxf(x, __shfl_xor(x, 2));
  x = fmaxf(x, __shfl_xor(x, 4));
  x = fmaxf(x, __shfl_xor(x, 8));
  return x;  // masks <=8 stay within each 16-lane half
}
__device__ __forceinline__ float rsum16(float x) {
  x += __shfl_xor(x, 1);
  x += __shfl_xor(x, 2);
  x += __shfl_xor(x, 4);
  x += __shfl_xor(x, 8);
  return x;
}

// ---------------------------------------------------------------------------
// Kernel 0: convert weights to bf16, transposed [n][k] so a WMMA B-fragment
// lane (column n, k-range half*16..half*16+15) is one contiguous 32B read.
// ---------------------------------------------------------------------------
__global__ __launch_bounds__(256) void k_wconv(const float* __restrict__ Wqkv,
                                               const float* __restrict__ Wgate,
                                               const float* __restrict__ Wout,
                                               __bf16* __restrict__ wcatT,
                                               __bf16* __restrict__ woutT) {
  int idx = blockIdx.x * 256 + threadIdx.x;
  if (idx < NCAT * C) {
    int n = idx >> 7, k = idx & 127;
    float v = (n < NQKV) ? Wqkv[k * NQKV + n] : Wgate[k * 128 + (n - NQKV)];
    wcatT[n * C + k] = (__bf16)v;
  } else {
    int j = idx - NCAT * C;
    if (j < C * C) {
      int n = j >> 7, k = j & 127;
      woutT[n * C + k] = (__bf16)Wout[k * C + n];
    }
  }
}

// ---------------------------------------------------------------------------
// Kernel 1: fused LayerNorm + (QKV | gate) projection via bf16 WMMA,
// plus pair-bias dot products. One 16-pixel tile per 128-thread block.
// ---------------------------------------------------------------------------
__global__ __launch_bounds__(128) void k_lnproj(
    const float* __restrict__ z, const float* __restrict__ lng,
    const float* __restrict__ lnb, const __bf16* __restrict__ wcatT,
    const float* __restrict__ Wpair, const float* __restrict__ bgate,
    __bf16* __restrict__ qbf, __bf16* __restrict__ kbf,
    __bf16* __restrict__ vT, __bf16* __restrict__ gatebf,
    float* __restrict__ pbias) {
  __shared__ float zf[16][C];
  __shared__ alignas(32) __bf16 znb[16][C + 8];  // pad: conflict-free A reads
  __shared__ float mu[16], rsg[16];

  const int p0 = blockIdx.x * 16;
  const int a0 = p0 / L;   // uniform: 16-pixel tile never crosses a row
  const int b20 = p0 % L;
  const int tid = threadIdx.x;

  if (tid == 0) __builtin_prefetch(wcatT, 0, 0);

  for (int e = tid; e < 16 * C; e += 128)
    zf[e >> 7][e & 127] = z[(size_t)(p0 + (e >> 7)) * C + (e & 127)];
  __syncthreads();

  {  // LN stats: 8 threads per pixel, shfl butterfly within 8-lane groups
    int p = tid >> 3, sub = tid & 7;
    float s = 0.f;
    for (int c = sub; c < C; c += 8) s += zf[p][c];
    s += __shfl_xor(s, 1); s += __shfl_xor(s, 2); s += __shfl_xor(s, 4);
    float m = s * (1.f / C);
    float v = 0.f;
    for (int c = sub; c < C; c += 8) { float d = zf[p][c] - m; v += d * d; }
    v += __shfl_xor(v, 1); v += __shfl_xor(v, 2); v += __shfl_xor(v, 4);
    if (sub == 0) { mu[p] = m; rsg[p] = rsqrtf(v * (1.f / C) + 1e-5f); }
  }
  __syncthreads();

  for (int e = tid; e < 16 * C; e += 128) {
    int r = e >> 7, c = e & 127;
    float zn = (zf[r][c] - mu[r]) * rsg[r] * lng[c] + lnb[c];
    znb[r][c] = (__bf16)zn;
    zf[r][c] = zn;  // keep fp32 zn for pair bias
  }
  __syncthreads();

  // pair bias: 16 pixels x 4 heads, fp32 dot over 128
  if (tid < 64) {
    int r = tid >> 2, h = tid & 3;
    float s = 0.f;
    for (int c = 0; c < C; ++c) s += zf[r][c] * Wpair[c * H + h];
    pbias[((size_t)h * L + a0) * L + b20 + r] = s;
  }

  const int lane = tid & 31;
  const int wave = __builtin_amdgcn_readfirstlane(tid >> 5);  // force SGPR
  const int m = lane & 15, half = lane >> 4;

  // Preload all four K-step A fragments (contiguous k = half*16 + s).
  v16bf A[4];
#pragma unroll
  for (int ks = 0; ks < 4; ++ks) {
    union { v16bf v; unsigned u[8]; } au;
#pragma unroll
    for (int vv = 0; vv < 8; ++vv) {
      int k = ks * 32 + half * 16 + vv * 2;
      au.u[vv] = *(const unsigned*)&znb[m][k];
    }
    A[ks] = au.v;
  }

  const float kscale = 0.17677669529663687f;  // 1/sqrt(32)
  for (int nt = wave; nt < 32; nt += 4) {
    const int ncol = nt * 16 + m;
    v8f acc = {};
#pragma unroll
    for (int ks = 0; ks < 4; ++ks) {
      v16bf B = *(const v16bf*)&wcatT[(size_t)ncol * C + ks * 32 + half * 16];
      acc = wmma_bf16(A[ks], B, acc);
    }
    // nt is SGPR-resident -> scalar branch selects q/k/v/gate per tile
    const int which = nt >> 3;
    const int w2 = (nt & 7) * 16 + m;  // 0..127 within the selected output
    if (which < 3) {
      const int h = w2 >> 5, d = w2 & 31;
      if (which == 2) {
        __bf16* dst = vT + ((((size_t)a0 * H + h) * D + d) * L + b20);
#pragma unroll
        for (int vv = 0; vv < 8; ++vv)
          dst[vv + half * 8] = (__bf16)acc[vv];  // contiguous: wide store
      } else {
        __bf16* dst = (which == 0 ? qbf : kbf) +
                      ((((size_t)a0 * H + h) * L + b20) * D + d);
        const float sc = (which == 0) ? 1.f : kscale;
#pragma unroll
        for (int vv = 0; vv < 8; ++vv)
          dst[(size_t)(vv + half * 8) * D] = (__bf16)(acc[vv] * sc);
      }
    } else {
      const float bg = bgate[w2];
      __bf16* dst = gatebf + (size_t)p0 * C + w2;
#pragma unroll
      for (int vv = 0; vv < 8; ++vv) {
        float gv = __builtin_amdgcn_rcpf(1.f + __expf(-(acc[vv] + bg)));
        dst[(size_t)(vv + half * 8) * C] = (__bf16)gv;
      }
    }
  }
}

// ---------------------------------------------------------------------------
// Kernel 2: flash-style row attention per (r,h). 256 threads = 8 waves,
// each wave owns 16-query tiles; online softmax; all matmuls are WMMA bf16.
// K tile staged via the Tensor Data Mover (TDM) with hardware LDS padding.
// ---------------------------------------------------------------------------
#define KPAD 48   // 32 data + 16 pad bf16 -> 96B rows, 32B-aligned B reads
#define VPAD 336  // 672B rows -> 32B-aligned B reads
__global__ __launch_bounds__(256) void k_attn(
    const __bf16* __restrict__ qbf, const __bf16* __restrict__ kbf,
    const __bf16* __restrict__ vT, const float* __restrict__ pbias,
    __bf16* __restrict__ attnbf) {
  __shared__ alignas(32) __bf16 Ks[L][KPAD];
  __shared__ alignas(32) __bf16 Vs[D][VPAD];
  __shared__ alignas(32) __bf16 pbuf[8][16][34];

  const int r = blockIdx.x >> 2, h = blockIdx.x & 3;
  const int tid = threadIdx.x;
  const __bf16* kb = kbf + (((size_t)r * H + h) * L) * D;
  const __bf16* vb = vT + (((size_t)r * H + h) * D) * L;

#if __has_builtin(__builtin_amdgcn_tensor_load_to_lds)
  if (tid < 32) {  // wave 0 issues one TDM DMA: [320][32] bf16 -> padded LDS
    unsigned lds_addr = (unsigned)(size_t)&Ks[0][0];
    unsigned long long ga = (unsigned long long)(size_t)kb;
    // D# group0: count=1 | lds_addr | global_addr[56:0] | type=2
    u32x4 g0 = {1u, lds_addr, (unsigned)(ga & 0xffffffffu),
                (unsigned)((ga >> 32) & 0x01ffffffu) | (2u << 30)};
    // D# group1: data_size=2B, pad_enable, pad_interval=16 DW (one 64B row),
    // pad_amount=8 DW (32B), tensor 32x320, tile 32x320, dim0 stride 32
    i32x8 g1 = {(1 << 16) | (1 << 20) | (3 << 22) | (7 << 25),
                (D << 16),        // tensor_dim0 = 32 (bits 79:48 low half)
                (L << 16),        // tensor_dim1 = 320 (low half)
                (D << 16),        // tile_dim0 = 32
                L,                // tile_dim1 = 320, tile_dim2 = 0
                D,                // tensor_dim0_stride = 32
                0, 0};
    i32x4 z4 = {0, 0, 0, 0};
#if __clang_major__ >= 23
    i32x8 z8 = {0, 0, 0, 0, 0, 0, 0, 0};
    __builtin_amdgcn_tensor_load_to_lds(g0, g1, z4, z4, z8, 0);
#else
    __builtin_amdgcn_tensor_load_to_lds(g0, g1, z4, z4, 0);
#endif
    __builtin_amdgcn_s_wait_tensorcnt(0);
  }
#else
  for (int e = tid; e < L * D; e += 256) Ks[e >> 5][e & 31] = kb[e];
#endif
  // V^T cooperative load, no div/mod in the addressing
  for (int rr = 0; rr < D; ++rr)
    for (int cc = tid; cc < L; cc += 256) Vs[rr][cc] = vb[rr * L + cc];
  __syncthreads();

  const int lane = tid & 31;
  const int wave = __builtin_amdgcn_readfirstlane(tid >> 5);  // force SGPR
  const int m = lane & 15, half = lane >> 4;
  const float* pb = pbias + (size_t)h * L * L;

  for (int qt = wave; qt < L / 16; qt += 8) {
    // Q A-fragment, per-lane 4B loads in contiguous k = half*16 + s layout
    const __bf16* qrow = qbf + ((((size_t)r * H + h) * L) + qt * 16 + m) * D;
    union { v16bf v; unsigned u[8]; } qa;
#pragma unroll
    for (int vv = 0; vv < 8; ++vv) {
      int k = half * 16 + vv * 2;
      qa.u[vv] = *(const unsigned*)&qrow[k];
    }

    v8f acc0 = {}, acc1 = {};
    float mrow[8], lrow[8];
#pragma unroll
    for (int vv = 0; vv < 8; ++vv) { mrow[vv] = -1e30f; lrow[vv] = 0.f; }

    for (int jb = 0; jb < L / 32; ++jb) {
      const int j0 = jb * 32;
      // scores: Q (16x32) x K^T (32x16) twice (j cols 0-15, 16-31)
      v16bf bk0 = *(const v16bf*)&Ks[j0 + m][half * 16];
      v16bf bk1 = *(const v16bf*)&Ks[j0 + 16 + m][half * 16];
      v8f s0 = {}, s1 = {};
      s0 = wmma_bf16(qa.v, bk0, s0);
      s1 = wmma_bf16(qa.v, bk1, s1);

#pragma unroll
      for (int vv = 0; vv < 8; ++vv) {
        int row = qt * 16 + vv + half * 8;
        float x0 = s0[vv] + pb[(size_t)row * L + j0 + m];
        float x1 = s1[vv] + pb[(size_t)row * L + j0 + 16 + m];
        float mb = rmax16(fmaxf(x0, x1));
        float mnew = fmaxf(mrow[vv], mb);
        float alpha = __expf(mrow[vv] - mnew);
        float p0e = __expf(x0 - mnew);
        float p1e = __expf(x1 - mnew);
        lrow[vv] = lrow[vv] * alpha + rsum16(p0e + p1e);
        mrow[vv] = mnew;
        acc0[vv] *= alpha;
        acc1[vv] *= alpha;
        // stage P (C layout -> LDS) for re-read in A layout
        pbuf[wave][vv + half * 8][m]      = (__bf16)p0e;
        pbuf[wave][vv + half * 8][m + 16] = (__bf16)p1e;
      }
      asm volatile("s_wait_dscnt 0" ::: "memory");
      union { v16bf v; unsigned u[8]; } pa;
#pragma unroll
      for (int vv = 0; vv < 8; ++vv) {
        int k = half * 16 + vv * 2;
        pa.u[vv] = *(const unsigned*)&pbuf[wave][m][k];
      }
      // out += P (16x32) x V (32x32 in two 16-col tiles)
      v16bf bv0 = *(const v16bf*)&Vs[m][j0 + half * 16];
      v16bf bv1 = *(const v16bf*)&Vs[16 + m][j0 + half * 16];
      acc0 = wmma_bf16(pa.v, bv0, acc0);
      acc1 = wmma_bf16(pa.v, bv1, acc1);
    }

#pragma unroll
    for (int vv = 0; vv < 8; ++vv) {
      int i = qt * 16 + vv + half * 8;
      float linv = __builtin_amdgcn_rcpf(lrow[vv]);
      size_t base = ((size_t)r * L + i) * (H * D) + h * D;
      attnbf[base + m]      = (__bf16)(acc0[vv] * linv);
      attnbf[base + 16 + m] = (__bf16)(acc1[vv] * linv);
    }
  }
}

// ---------------------------------------------------------------------------
// Kernel 3: (gate * out) @ Wout + bout via bf16 WMMA. 16-pixel tile/block,
// 8 waves each own one 16-column slab of the 128 outputs.
// ---------------------------------------------------------------------------
__global__ __launch_bounds__(256) void k_out(
    const __bf16* __restrict__ gatebf, const __bf16* __restrict__ attnbf,
    const __bf16* __restrict__ woutT, const float* __restrict__ bout,
    float* __restrict__ out) {
  __shared__ alignas(32) __bf16 ab[16][C + 8];
  const int p0 = blockIdx.x * 16;
  const int tid = threadIdx.x;

  for (int e = tid; e < 16 * C; e += 256) {
    int rr = e >> 7, cc = e & 127;
    size_t idx = (size_t)(p0 + rr) * C + cc;
    ab[rr][cc] = (__bf16)((float)gatebf[idx] * (float)attnbf[idx]);
  }
  __syncthreads();

  const int lane = tid & 31;
  const int wave = __builtin_amdgcn_readfirstlane(tid >> 5);  // force SGPR
  const int m = lane & 15, half = lane >> 4;
  const int ncol = wave * 16 + m;

  v16bf A[4];
#pragma unroll
  for (int ks = 0; ks < 4; ++ks) {
    union { v16bf v; unsigned u[8]; } au;
#pragma unroll
    for (int vv = 0; vv < 8; ++vv) {
      int k = ks * 32 + half * 16 + vv * 2;
      au.u[vv] = *(const unsigned*)&ab[m][k];
    }
    A[ks] = au.v;
  }

  v8f acc = {};
#pragma unroll
  for (int ks = 0; ks < 4; ++ks) {
    v16bf B = *(const v16bf*)&woutT[(size_t)ncol * C + ks * 32 + half * 16];
    acc = wmma_bf16(A[ks], B, acc);
  }

  const float bo = bout[ncol];
#pragma unroll
  for (int vv = 0; vv < 8; ++vv)
    out[(size_t)(p0 + vv + half * 8) * C + ncol] = acc[vv] + bo;
}

// ---------------------------------------------------------------------------
extern "C" void kernel_launch(void* const* d_in, const int* in_sizes, int n_in,
                              void* d_out, int out_size, void* d_ws,
                              size_t ws_size, hipStream_t stream) {
  const float* z     = (const float*)d_in[0];
  const float* ln_g  = (const float*)d_in[1];
  const float* ln_b  = (const float*)d_in[2];
  const float* Wqkv  = (const float*)d_in[3];
  const float* Wpair = (const float*)d_in[4];
  const float* Wgate = (const float*)d_in[5];
  const float* bgate = (const float*)d_in[6];
  const float* Wout  = (const float*)d_in[7];
  const float* bout  = (const float*)d_in[8];
  float* out = (float*)d_out;

  char* ws = (char*)d_ws;
  const size_t QKV_BYTES = (size_t)L * L * H * D * sizeof(__bf16);  // 26,214,400
  const size_t O_WCAT = 0;                                    // 512*128*2
  const size_t O_WOUT = O_WCAT + (size_t)NCAT * C * 2;        // +131072
  const size_t O_Q    = O_WOUT + (size_t)C * C * 2;           // +32768
  const size_t O_K    = O_Q + QKV_BYTES;
  const size_t O_VT   = O_K + QKV_BYTES;
  const size_t O_PB   = O_VT + QKV_BYTES;
  const size_t O_GATE = O_PB + (size_t)H * L * L * sizeof(float);
  const size_t O_ATTN = O_GATE + QKV_BYTES;

  __bf16* wcatT  = (__bf16*)(ws + O_WCAT);
  __bf16* woutT  = (__bf16*)(ws + O_WOUT);
  __bf16* qbf    = (__bf16*)(ws + O_Q);
  __bf16* kbf    = (__bf16*)(ws + O_K);
  __bf16* vT     = (__bf16*)(ws + O_VT);
  float*  pbias  = (float*)(ws + O_PB);
  __bf16* gatebf = (__bf16*)(ws + O_GATE);
  __bf16* attnbf = (__bf16*)(ws + O_ATTN);

  const int nTiles = L * L / 16;  // 6400 pixel tiles

  k_wconv<<<(NCAT * C + C * C + 255) / 256, 256, 0, stream>>>(
      Wqkv, Wgate, Wout, wcatT, woutT);
  k_lnproj<<<nTiles, 128, 0, stream>>>(z, ln_g, ln_b, wcatT, Wpair, bgate, qbf,
                                       kbf, vT, gatebf, pbias);
  k_attn<<<L * H, 256, 0, stream>>>(qbf, kbf, vT, pbias, attnbf);
  k_out<<<nTiles, 256, 0, stream>>>(gatebf, attnbf, woutT, bout, out);
}
